// IntraSentenceGNN_58884001628475
// MI455X (gfx1250) — compile-verified, matrix operand order checked by reading, not run on hardware
//
#include <hip/hip_runtime.h>
#include <hip/hip_bf16.h>
#include <stdint.h>

// ---------------------------------------------------------------------------
// IntraSentenceGNN for MI455X (gfx1250).
// B=32768, D=H=1024, HEADS=8, DH=128.  wave32, WMMA f32_16x16x32_bf16,
// Tensor Data Mover (TDM) staging for the GAT-layer GEMMs.
// ---------------------------------------------------------------------------

typedef __bf16 bf16;
typedef __attribute__((ext_vector_type(8)))  __bf16 v8bf;
typedef __attribute__((ext_vector_type(16))) __bf16 v16bf;
typedef __attribute__((ext_vector_type(8)))  float  v8f;
typedef __attribute__((ext_vector_type(4)))  unsigned int v4u;
typedef __attribute__((ext_vector_type(8)))  int v8i;
typedef __attribute__((ext_vector_type(4)))  int v4i;

#define D_    1024
#define B_    32768
#define ROWS_ (B_ * 3)
#define BK_   32
#define LDA_  40      // padded LDS row stride in bf16 elements (80B, 16B aligned)
#define GLD_  260     // padded f32 LDS stride for gl/gr tile

#if defined(__has_builtin)
#  if __has_builtin(__builtin_amdgcn_tensor_load_to_lds)
#    define HAVE_TDM 1
#  else
#    define HAVE_TDM 0
#  endif
#else
#  define HAVE_TDM 0
#endif

// --- WMMA fragment loader: ISA 16-bit A/B layout ---------------------------
// lane (0..31): row/col = lane&15, khalf = lane>>4.
// v16 elements 0..7  = K (khalf*8 .. +7), elements 8..15 = K (16+khalf*8 .. +7)
__device__ __forceinline__ v16bf ld_frag(const bf16* rowbase, int khalf) {
  const v8bf lo = *(const v8bf*)(rowbase + khalf * 8);
  const v8bf hi = *(const v8bf*)(rowbase + 16 + khalf * 8);
  return __builtin_shufflevector(lo, hi, 0,1,2,3,4,5,6,7,8,9,10,11,12,13,14,15);
}

__device__ __forceinline__ v8f wmma_bf16(v16bf a, v16bf b, v8f c) {
  return __builtin_amdgcn_wmma_f32_16x16x32_bf16(false, a, false, b, (short)0,
                                                 c, false, false);
}

#if HAVE_TDM
// ---------------------------------------------------------------------------
// TDM 2-D tile load: tileRows x 32 bf16 elements, source pitch 1024 elements,
// LDS pitch LDA_=40 elements via pad feature (pad 4 DWORDs every 16 DWORDs).
// Descriptor packed per CDNA5 ISA D# group0/group1 layout (ISA p.152-156).
// Rows >= tensorRows read as zero (hardware OOB zero-fill).
// ---------------------------------------------------------------------------
__device__ __forceinline__ void tdm_load_2d(unsigned ldsAddr, const void* gptr,
                                            unsigned tensorRows,
                                            unsigned tileRows) {
  const unsigned long long ga = (unsigned long long)(size_t)gptr;
  v4u g0;
  g0.x = 1u;                                   // count=1, user mode, no gather
  g0.y = ldsAddr;                              // lds_addr [63:32]
  g0.z = (unsigned)ga;                         // global_addr [95:64]
  g0.w = (unsigned)(ga >> 32) | (2u << 30);    // global_addr hi | type=2
  v8i g1;
  //      data_size=2B   pad_en     interval=16dw  amount=4dw
  g1[0] = (int)((1u << 16) | (1u << 20) | (3u << 22) | (3u << 25));
  g1[1] = (int)(1024u << 16);                  // tensor_dim0 = 1024 (lo16)
  g1[2] = (int)((tensorRows & 0xffffu) << 16); // tensor_dim1 lo16
  g1[3] = (int)((tensorRows >> 16) | (32u << 16)); // tensor_dim1 hi | tile_dim0=32
  g1[4] = (int)tileRows;                       // tile_dim1 (tile_dim2=0)
  g1[5] = (int)1024;                           // tensor_dim0_stride = 1024
  g1[6] = 0;
  g1[7] = 0;
  const v4i gz = {0, 0, 0, 0};
#if __has_include(<hip/amd_detail/amd_gfx1250_TDM.h>)
  const v8i gz8 = {0, 0, 0, 0, 0, 0, 0, 0};
  __builtin_amdgcn_tensor_load_to_lds(g0, g1, gz, gz, gz8, 0);
#else
  __builtin_amdgcn_tensor_load_to_lds(g0, g1, gz, gz, 0);
#endif
}
#endif  // HAVE_TDM

// ---------------------------------------------------------------------------
// Weight convert: f32 [K][N] -> bf16 transposed [N][K]  (LDS-tiled transpose)
// ---------------------------------------------------------------------------
__global__ __launch_bounds__(256) void k_wconv(const float* __restrict__ in,
                                               bf16* __restrict__ out) {
  __shared__ float tile[32][33];
  const int bx = blockIdx.x * 32;   // n base
  const int by = blockIdx.y * 32;   // k base
  const int tx = threadIdx.x, ty = threadIdx.y;   // 32 x 8
  #pragma unroll
  for (int i = 0; i < 32; i += 8)
    tile[ty + i][tx] = in[(size_t)(by + ty + i) * D_ + bx + tx];
  __syncthreads();
  #pragma unroll
  for (int i = 0; i < 32; i += 8)
    out[(size_t)(bx + ty + i) * D_ + by + tx] = (bf16)tile[tx][ty + i];
}

// ---------------------------------------------------------------------------
// Projection GEMM: x_mod = in[32768,1024] @ W + bias -> bf16 x[(b*3+z)*1024+n]
// Tile 128x128, BK=32.  8 waves as 4(M) x 2(N); wave tile 32x64 (2x4 WMMA).
// (Manual staging: must convert f32 -> bf16, which TDM cannot do.)
// ---------------------------------------------------------------------------
__global__ __launch_bounds__(256) void k_proj(const float* __restrict__ A,
                                              const bf16* __restrict__ WT,
                                              const float* __restrict__ bias,
                                              bf16* __restrict__ xout, int z) {
  __shared__ bf16 ldsA[128 * LDA_];
  __shared__ bf16 ldsB[128 * LDA_];
  const int tid = threadIdx.x;
  const int lane = tid & 31, wave = tid >> 5;
  const int wm = wave >> 1, wn = wave & 1;
  const int m0 = blockIdx.y * 128, n0 = blockIdx.x * 128;

  v8f zero = {};
  v8f acc[2][4];
  #pragma unroll
  for (int mt = 0; mt < 2; ++mt)
    #pragma unroll
    for (int nt = 0; nt < 4; ++nt) acc[mt][nt] = zero;

  const int sr = tid >> 1, sh = tid & 1;
  for (int k0 = 0; k0 < D_; k0 += BK_) {
    __syncthreads();
    {   // stage A: convert f32 -> bf16 (16 elems / thread)
      const float* src = A + (size_t)(m0 + sr) * D_ + k0 + sh * 16;
      bf16* dst = ldsA + sr * LDA_ + sh * 16;
      #pragma unroll
      for (int i = 0; i < 16; i += 4) {
        float4 f = *(const float4*)(src + i);
        dst[i + 0] = (bf16)f.x; dst[i + 1] = (bf16)f.y;
        dst[i + 2] = (bf16)f.z; dst[i + 3] = (bf16)f.w;
      }
    }
    {   // stage B from transposed bf16 weights (coalesced along K)
      const bf16* src = WT + (size_t)(n0 + sr) * D_ + k0 + sh * 16;
      if (k0 + BK_ < D_) __builtin_prefetch(src + BK_, 0, 1);
      v8bf* dst = (v8bf*)(ldsB + sr * LDA_ + sh * 16);
      dst[0] = *(const v8bf*)(src);
      dst[1] = *(const v8bf*)(src + 8);
    }
    __syncthreads();
    const int khalf = lane >> 4, l16 = lane & 15;
    v16bf af[2];
    #pragma unroll
    for (int mt = 0; mt < 2; ++mt)
      af[mt] = ld_frag(ldsA + (wm * 32 + mt * 16 + l16) * LDA_, khalf);
    #pragma unroll
    for (int nt = 0; nt < 4; ++nt) {
      v16bf bfr = ld_frag(ldsB + (wn * 64 + nt * 16 + l16) * LDA_, khalf);
      #pragma unroll
      for (int mt = 0; mt < 2; ++mt)
        acc[mt][nt] = wmma_bf16(af[mt], bfr, acc[mt][nt]);
    }
  }
  // epilogue: C layout -> row = lanehalf*8 + r, col = lane&15
  const int l16 = lane & 15, lh = lane >> 4;
  #pragma unroll
  for (int mt = 0; mt < 2; ++mt)
    #pragma unroll
    for (int nt = 0; nt < 4; ++nt) {
      const int col = n0 + wn * 64 + nt * 16 + l16;
      const float bcol = bias[col];
      const int rbase = m0 + wm * 32 + mt * 16 + lh * 8;
      #pragma unroll
      for (int r = 0; r < 8; ++r) {
        float v = acc[mt][nt][r] + bcol;
        xout[((size_t)(rbase + r) * 3 + z) * D_ + col] = (bf16)v;
      }
    }
}

// ---------------------------------------------------------------------------
// Shared GEMM body for GAT layers: tile M=128 (120 used = 40 samples x 3
// nodes) x N=256 ([gl 128 | gr 128]), K=1024.  Result -> ldsGL f32.
// 8 waves as 4(M) x 2(N); wave tile 32x128 (2x8 WMMA accumulators).
// Staging: TDM tensor_load_to_lds (waves 0/1/2 issue A / B-left / B-right
// descriptors, wait on their TENSORcnt, barrier publishes to all waves).
// ---------------------------------------------------------------------------
__device__ __forceinline__ void gat_gemm(const bf16* __restrict__ xsrc,
                                         int grow0, int nValidRows,
                                         const bf16* __restrict__ BL,
                                         const bf16* __restrict__ BR,
                                         bf16* ldsA, bf16* ldsB, float* ldsGL) {
  const int tid = threadIdx.x;
  const int lane = tid & 31, wave = tid >> 5;
  const int wm = wave >> 1, wn = wave & 1;

  v8f zero = {};
  v8f acc[2][8];
  #pragma unroll
  for (int mt = 0; mt < 2; ++mt)
    #pragma unroll
    for (int nt = 0; nt < 8; ++nt) acc[mt][nt] = zero;

#if HAVE_TDM
  const unsigned ldsAAddr = (unsigned)(size_t)(void*)ldsA;
  const unsigned ldsBAddr = (unsigned)(size_t)(void*)ldsB;
  // Rows past the end of x zero-fill in hardware; rows 120..127 of interior
  // tiles load real neighbor rows but land in GL rows the epilogue never
  // reads (only b < MBv is consumed).
  const unsigned rowsA = (unsigned)(ROWS_ - grow0);
  (void)nValidRows;
#else
  const int sr = tid >> 1, sh = tid & 1;
#endif

  for (int k0 = 0; k0 < D_; k0 += BK_) {
    __syncthreads();
#if HAVE_TDM
    if (wave == 0)
      tdm_load_2d(ldsAAddr, xsrc + (size_t)grow0 * D_ + k0, rowsA, 128);
    else if (wave == 1)
      tdm_load_2d(ldsBAddr, BL + k0, 128, 128);
    else if (wave == 2)
      tdm_load_2d(ldsBAddr + 128 * LDA_ * 2, BR + k0, 128, 128);
    if (wave < 3) __builtin_amdgcn_s_wait_tensorcnt(0);
#else
    {   // stage A (zero-pad rows >= nValidRows)
      v8bf lo = {}, hi = {};
      if (sr < nValidRows) {
        const bf16* src = xsrc + (size_t)(grow0 + sr) * D_ + k0 + sh * 16;
        lo = *(const v8bf*)src;
        hi = *(const v8bf*)(src + 8);
      }
      v8bf* dst = (v8bf*)(ldsA + sr * LDA_ + sh * 16);
      dst[0] = lo; dst[1] = hi;
    }
    {   // stage B: 256 rows (cols), 32 bf16 (64B) each thread
      const bf16* src = (tid < 128) ? (BL + (size_t)tid * D_ + k0)
                                    : (BR + (size_t)(tid - 128) * D_ + k0);
      if (k0 + BK_ < D_) __builtin_prefetch(src + BK_, 0, 1);
      v8bf* dst = (v8bf*)(ldsB + tid * LDA_);
      const v8bf* s8 = (const v8bf*)src;
      dst[0] = s8[0]; dst[1] = s8[1]; dst[2] = s8[2]; dst[3] = s8[3];
    }
#endif
    __syncthreads();
    const int khalf = lane >> 4, l16 = lane & 15;
    v16bf af[2];
    #pragma unroll
    for (int mt = 0; mt < 2; ++mt)
      af[mt] = ld_frag(ldsA + (wm * 32 + mt * 16 + l16) * LDA_, khalf);
    #pragma unroll
    for (int nt = 0; nt < 8; ++nt) {
      v16bf bfr = ld_frag(ldsB + (wn * 128 + nt * 16 + l16) * LDA_, khalf);
      #pragma unroll
      for (int mt = 0; mt < 2; ++mt)
        acc[mt][nt] = wmma_bf16(af[mt], bfr, acc[mt][nt]);
    }
  }
  // spill accumulators to LDS for the cross-wave attention epilogue
  __syncthreads();
  const int l16 = lane & 15, lh = lane >> 4;
  #pragma unroll
  for (int mt = 0; mt < 2; ++mt)
    #pragma unroll
    for (int nt = 0; nt < 8; ++nt) {
      const int row = wm * 32 + mt * 16 + lh * 8;
      const int col = wn * 128 + nt * 16 + l16;
      #pragma unroll
      for (int r = 0; r < 8; ++r)
        ldsGL[(row + r) * GLD_ + col] = acc[mt][nt][r];
    }
  __syncthreads();
}

// LDS layout for GAT kernels (dynamic, 165792 B total):
//   [0,10240)        ldsA   128x40 bf16
//   [10240,30720)    ldsB   256x40 bf16
//   [30720,163840)   ldsGL  128x260 f32   (gl cols 0..127 | gr cols 128..255)
//   [163840,165280)  ldsE   40x9 f32
//   [165280,165792)  ldsAtt 128 f32
#define SHM_GAT 165792

// ---------------------------------------------------------------------------
// GAT layer 1, fully fused per (head, 40-sample tile).
// ---------------------------------------------------------------------------
__global__ __launch_bounds__(256) void k_gat1(const bf16* __restrict__ x,
                                              const bf16* __restrict__ Wl1T,
                                              const bf16* __restrict__ Wr1T,
                                              const float* __restrict__ att1,
                                              const float* __restrict__ b1,
                                              bf16* __restrict__ x1) {
  extern __shared__ char smem[];
  bf16*  ldsA  = (bf16*)smem;
  bf16*  ldsB  = (bf16*)(smem + 10240);
  float* ldsGL = (float*)(smem + 30720);
  float* ldsE  = (float*)(smem + 163840);
  float* ldsAtt= (float*)(smem + 165280);

  const int h = blockIdx.x, bt = blockIdx.y, tid = threadIdx.x;
  const int row0 = bt * 120;
  int MBv = B_ - bt * 40; if (MBv > 40) MBv = 40;

  if (tid < 128) ldsAtt[tid] = att1[h * 128 + tid];

  gat_gemm(x, row0, MBv * 3,
           Wl1T + (size_t)h * 128 * D_, Wr1T + (size_t)h * 128 * D_,
           ldsA, ldsB, ldsGL);

  // e[b][i][j] = sum_d leakyrelu(gl[b,j,d] + gr[b,i,d], 0.2) * att1[h,d]
  for (int idx = tid; idx < MBv * 9; idx += 256) {
    const int b = idx / 9, p = idx - b * 9, i = p / 3, j = p - i * 3;
    const float* gl = ldsGL + (b * 3 + j) * GLD_;         // cols 0..127
    const float* gr = ldsGL + (b * 3 + i) * GLD_ + 128;   // cols 128..255
    float s = 0.f;
    #pragma unroll 8
    for (int d = 0; d < 128; ++d) {
      float v = gl[d] + gr[d];
      v = (v > 0.f) ? v : 0.2f * v;
      s += v * ldsAtt[d];
    }
    ldsE[idx] = s;
  }
  __syncthreads();

  // softmax over j, aggregate, + b1, ELU, store bf16
  for (int idx = tid; idx < MBv * 384; idx += 256) {
    const int b = idx / 384, rem = idx - b * 384;
    const int i = rem >> 7, d = rem & 127;
    const float e0 = ldsE[b * 9 + i * 3 + 0];
    const float e1 = ldsE[b * 9 + i * 3 + 1];
    const float e2 = ldsE[b * 9 + i * 3 + 2];
    const float m = fmaxf(e0, fmaxf(e1, e2));
    const float a0 = __expf(e0 - m), a1 = __expf(e1 - m), a2 = __expf(e2 - m);
    const float inv = 1.f / (a0 + a1 + a2);
    float o = (a0 * ldsGL[(b * 3 + 0) * GLD_ + d] +
               a1 * ldsGL[(b * 3 + 1) * GLD_ + d] +
               a2 * ldsGL[(b * 3 + 2) * GLD_ + d]) * inv;
    o += b1[h * 128 + d];
    o = (o > 0.f) ? o : (__expf(o) - 1.f);          // ELU
    x1[(size_t)(row0 + b * 3 + i) * D_ + h * 128 + d] = (bf16)o;
  }
}

// ---------------------------------------------------------------------------
// GAT layer 2 pass A: per (column-chunk c, 40-sample tile): gl2/gr2 slices,
// partial scores ePart[b][i][j][c] (deterministic, no atomics), gl2 -> bf16.
// ---------------------------------------------------------------------------
__global__ __launch_bounds__(256) void k_gat2a(const bf16* __restrict__ x1,
                                               const bf16* __restrict__ Wl2T,
                                               const bf16* __restrict__ Wr2T,
                                               const float* __restrict__ att2,
                                               float* __restrict__ ePart,
                                               bf16* __restrict__ gl2) {
  extern __shared__ char smem[];
  bf16*  ldsA  = (bf16*)smem;
  bf16*  ldsB  = (bf16*)(smem + 10240);
  float* ldsGL = (float*)(smem + 30720);
  float* ldsAtt= (float*)(smem + 165280);

  const int c = blockIdx.x, bt = blockIdx.y, tid = threadIdx.x;
  const int row0 = bt * 120;
  int MBv = B_ - bt * 40; if (MBv > 40) MBv = 40;

  if (tid < 128) ldsAtt[tid] = att2[c * 128 + tid];

  gat_gemm(x1, row0, MBv * 3,
           Wl2T + (size_t)c * 128 * D_, Wr2T + (size_t)c * 128 * D_,
           ldsA, ldsB, ldsGL);

  for (int idx = tid; idx < MBv * 9; idx += 256) {
    const int b = idx / 9, p = idx - b * 9, i = p / 3, j = p - i * 3;
    const float* gl = ldsGL + (b * 3 + j) * GLD_;
    const float* gr = ldsGL + (b * 3 + i) * GLD_ + 128;
    float s = 0.f;
    #pragma unroll 8
    for (int d = 0; d < 128; ++d) {
      float v = gl[d] + gr[d];
      v = (v > 0.f) ? v : 0.2f * v;
      s += v * ldsAtt[d];
    }
    ePart[((size_t)(bt * 40 + b) * 9 + p) * 8 + c] = s;
  }
  for (int idx = tid; idx < MBv * 384; idx += 256) {
    const int b = idx / 384, rem = idx - b * 384;
    const int i = rem >> 7, d = rem & 127;
    gl2[(size_t)(row0 + b * 3 + i) * D_ + c * 128 + d] =
        (bf16)ldsGL[(b * 3 + i) * GLD_ + d];
  }
}

// ---------------------------------------------------------------------------
// GAT layer 2 pass B: reduce ePart, softmax, out = sum_j w_j*gl2_j + b2
// with w_j = (1/3) sum_i alpha_ij  (mean over nodes folded in).
// ---------------------------------------------------------------------------
__global__ __launch_bounds__(256) void k_gat2b(const float* __restrict__ ePart,
                                               const bf16* __restrict__ gl2,
                                               const float* __restrict__ b2,
                                               float* __restrict__ out) {
  __shared__ float eS[9];
  __shared__ float w[3];
  const int b = blockIdx.x, tid = threadIdx.x;
  if (tid < 9) {
    const float* p = ePart + ((size_t)b * 9 + tid) * 8;
    float s = 0.f;
    #pragma unroll
    for (int c = 0; c < 8; ++c) s += p[c];
    eS[tid] = s;
  }
  __syncthreads();
  if (tid == 0) {
    float ww0 = 0.f, ww1 = 0.f, ww2 = 0.f;
    #pragma unroll
    for (int i = 0; i < 3; ++i) {
      const float e0 = eS[i * 3], e1 = eS[i * 3 + 1], e2 = eS[i * 3 + 2];
      const float m = fmaxf(e0, fmaxf(e1, e2));
      const float a0 = __expf(e0 - m), a1 = __expf(e1 - m), a2 = __expf(e2 - m);
      const float inv = 1.f / (a0 + a1 + a2);
      ww0 += a0 * inv; ww1 += a1 * inv; ww2 += a2 * inv;
    }
    w[0] = ww0 * (1.f / 3.f); w[1] = ww1 * (1.f / 3.f); w[2] = ww2 * (1.f / 3.f);
  }
  __syncthreads();
  const float w0 = w[0], w1 = w[1], w2 = w[2];
  const bf16* g = gl2 + (size_t)b * 3 * D_;
  for (int d = tid; d < D_; d += 256)
    out[(size_t)b * D_ + d] = w0 * (float)g[d] + w1 * (float)g[D_ + d] +
                              w2 * (float)g[2 * D_ + d] + b2[d];
}

// ---------------------------------------------------------------------------
extern "C" void kernel_launch(void* const* d_in, const int* in_sizes, int n_in,
                              void* d_out, int out_size, void* d_ws,
                              size_t ws_size, hipStream_t stream) {
  (void)in_sizes; (void)n_in; (void)out_size; (void)ws_size;
  const float* text  = (const float*)d_in[0];
  const float* audio = (const float*)d_in[1];
  const float* video = (const float*)d_in[2];
  const float* Wt  = (const float*)d_in[3];
  const float* bt  = (const float*)d_in[4];
  const float* Wa  = (const float*)d_in[5];
  const float* ba  = (const float*)d_in[6];
  const float* Wv  = (const float*)d_in[7];
  const float* bv  = (const float*)d_in[8];
  const float* Wl1 = (const float*)d_in[9];
  const float* Wr1 = (const float*)d_in[10];
  const float* att1= (const float*)d_in[11];
  const float* b1  = (const float*)d_in[12];
  const float* Wl2 = (const float*)d_in[13];
  const float* Wr2 = (const float*)d_in[14];
  const float* att2= (const float*)d_in[15];
  const float* b2  = (const float*)d_in[16];

  char* ws = (char*)d_ws;
  const size_t WB = (size_t)D_ * D_ * sizeof(bf16);        // 2 MiB / weight
  const size_t XB = (size_t)ROWS_ * D_ * sizeof(bf16);     // 192 MiB
  bf16* WtT  = (bf16*)(ws + 0 * WB);
  bf16* WaT  = (bf16*)(ws + 1 * WB);
  bf16* WvT  = (bf16*)(ws + 2 * WB);
  bf16* Wl1T = (bf16*)(ws + 3 * WB);
  bf16* Wr1T = (bf16*)(ws + 4 * WB);
  bf16* Wl2T = (bf16*)(ws + 5 * WB);
  bf16* Wr2T = (bf16*)(ws + 6 * WB);
  bf16* xbuf   = (bf16*)(ws + 7 * WB);
  bf16* x1buf  = (bf16*)(ws + 7 * WB + XB);
  float* ePart = (float*)(ws + 7 * WB + 2 * XB);           // 32768*9*8 f32
  bf16* gl2buf = xbuf;   // x dead after k_gat1 -> reuse for gl2

  const dim3 tb(32, 8), tg(32, 32);
  k_wconv<<<tg, tb, 0, stream>>>(Wt,  WtT);
  k_wconv<<<tg, tb, 0, stream>>>(Wa,  WaT);
  k_wconv<<<tg, tb, 0, stream>>>(Wv,  WvT);
  k_wconv<<<tg, tb, 0, stream>>>(Wl1, Wl1T);
  k_wconv<<<tg, tb, 0, stream>>>(Wr1, Wr1T);
  k_wconv<<<tg, tb, 0, stream>>>(Wl2, Wl2T);
  k_wconv<<<tg, tb, 0, stream>>>(Wr2, Wr2T);

  k_proj<<<dim3(8, 256), 256, 0, stream>>>(text,  WtT, bt, xbuf, 0);
  k_proj<<<dim3(8, 256), 256, 0, stream>>>(audio, WaT, ba, xbuf, 1);
  k_proj<<<dim3(8, 256), 256, 0, stream>>>(video, WvT, bv, xbuf, 2);

  const int NBT = (B_ + 39) / 40;   // 820 batch tiles of 40 samples
  k_gat1 <<<dim3(8, NBT), 256, SHM_GAT, stream>>>(xbuf, Wl1T, Wr1T, att1, b1,
                                                  x1buf);
  k_gat2a<<<dim3(8, NBT), 256, SHM_GAT, stream>>>(x1buf, Wl2T, Wr2T, att2,
                                                  ePart, gl2buf);
  k_gat2b<<<dim3(B_), 256, 0, stream>>>(ePart, gl2buf, b2, (float*)d_out);
}
